// TBInnerQuadLinearAP_43645457662667
// MI455X (gfx1250) — compile-verified, matrix operand order checked by reading, not run on hardware
//
#include <hip/hip_runtime.h>

typedef __attribute__((ext_vector_type(16))) _Float16 v16h;
typedef __attribute__((ext_vector_type(8)))  float    v8f;

#define SIGMA_F 0.1f
#define RHO_F   10.0f
#define BBDIM   64      // B*B
#define FDIM    64
#define ROWS    (BBDIM * FDIM)   // 4096
#define WPB     8                // waves per block (wave32)

// One wave32 per (b1,b2,i) row.
// PH[j] = sum_k p[k]*(s[k]>s[j])        via v_wmma_f32_16x16x32_f16 (0/1 exact in f16, f32 accum;
//                                        diagonal k==j self-masks since s[j]>s[j] is false)
// QN[j] = sum_{k!=j} w[k]*quad_linear(s[k]-s[j]),  w[k] = n[k]*(1-p[k])   (f32 VALU, branchless)
// rowval = ( sum_j p[j]*(1+rho*PH)/(1+rho*PH+QN) ) / ( sum_j p[j] )
__global__ __launch_bounds__(WPB * 32)
void tb_ap_rows_kernel(const float* __restrict__ sim,
                       const float* __restrict__ pos,
                       const float* __restrict__ neg,
                       float* __restrict__ rowval)
{
    __shared__ float sh[WPB][4][FDIM];   // per-wave: s, p, w, ph
    const int lane = threadIdx.x & 31;
    const int wave = threadIdx.x >> 5;
    const int row  = blockIdx.x * WPB + wave;

    float* s  = &sh[wave][0][0];
    float* p  = &sh[wave][1][0];
    float* w  = &sh[wave][2][0];
    float* ph = &sh[wave][3][0];

    const float* S = sim + (size_t)row * FDIM;
    const float* P = pos + (size_t)row * FDIM;
    const float* N = neg + (size_t)row * FDIM;

    {
        const float p0 = P[lane], p1 = P[lane + 32];
        const float n0 = N[lane], n1 = N[lane + 32];
        s[lane] = S[lane];          s[lane + 32] = S[lane + 32];
        p[lane] = p0;               p[lane + 32] = p1;
        w[lane] = n0 * (1.0f - p0); w[lane + 32] = n1 * (1.0f - p1);
    }
    __syncthreads();

    // ---------- exact Heaviside positive-count via WMMA ----------
    const int m = lane & 15;    // M index within 16x16 tile (A layout, 16-bit)
    const int g = lane >> 4;    // lane group selects K sub-range

    // Hoist this lane's 32 K-side operands out of the j-tile loop.
    // q = chunk*16 + h  ->  k = chunk*32 + h + 8*g + (h>=8 ? 8 : 0)
    float    skv[32];
    _Float16 pkv[32];
    #pragma unroll
    for (int q = 0; q < 32; ++q) {
        const int chunk = q >> 4;
        const int h     = q & 15;
        const int k     = chunk * 32 + h + 8 * g + ((h >= 8) ? 8 : 0);
        skv[q] = s[k];
        pkv[q] = (_Float16)p[k];
    }

    v16h onesv;
    #pragma unroll
    for (int h = 0; h < 16; ++h) onesv[h] = (_Float16)1.0f;
    const _Float16 zeroh = (_Float16)0.0f;

    #pragma unroll
    for (int t = 0; t < 4; ++t) {            // j-tile: j = t*16 + m
        const float sj = s[t * 16 + m];
        v8f c = {};
        #pragma unroll
        for (int chunk = 0; chunk < 2; ++chunk) {   // K = 64 in two K=32 steps
            v16h a;
            #pragma unroll
            for (int h = 0; h < 16; ++h) {
                const int q = chunk * 16 + h;
                a[h] = (skv[q] > sj) ? pkv[q] : zeroh;   // v_cmp + v_cndmask, branchless
            }
            c = __builtin_amdgcn_wmma_f32_16x16x32_f16(
                    false, a, false, onesv, (short)0, c, false, false);
        }
        // C/D layout: vgpr v, lanes 0-15 -> (M=v, N=lane), lanes 16-31 -> (M=v+8, N=lane-16).
        // B was all-ones so every N column equals the row sum; N==0 lanes scatter it.
        if (m == 0) {
            #pragma unroll
            for (int vv = 0; vv < 8; ++vv) ph[t * 16 + vv + 8 * g] = c[vv];
        }
    }
    __syncthreads();

    // ---------- quad_linear part (f32, branchless), both j's share one k sweep ----------
    const int   j0  = lane,      j1  = lane + 32;
    const float sj0 = s[j0],     sj1 = s[j1];
    float qn0 = 0.0f, qn1 = 0.0f;
    for (int k = 0; k < FDIM; ++k) {
        const float sk = s[k];
        const float wk = w[k];

        const float d0 = sk - sj0;
        const float u0 = d0 / SIGMA_F;
        const float q0 = (d0 > 0.0f) ? (2.0f * u0 + 1.0f)
                       : ((d0 >= -SIGMA_F) ? (u0 * u0 + 2.0f * u0 + 1.0f) : 0.0f);
        qn0 += (k != j0) ? wk * q0 : 0.0f;

        const float d1 = sk - sj1;
        const float u1 = d1 / SIGMA_F;
        const float q1 = (d1 > 0.0f) ? (2.0f * u1 + 1.0f)
                       : ((d1 >= -SIGMA_F) ? (u1 * u1 + 2.0f * u1 + 1.0f) : 0.0f);
        qn1 += (k != j1) ? wk * q1 : 0.0f;
    }

    float contrib, pcnt;
    {
        const float posr0 = 1.0f + RHO_F * ph[j0];
        const float posr1 = 1.0f + RHO_F * ph[j1];
        const float pj0 = p[j0], pj1 = p[j1];
        contrib = pj0 * (posr0 / (posr0 + qn0)) + pj1 * (posr1 / (posr1 + qn1));
        pcnt    = pj0 + pj1;
    }

    // wave32 reduction
    #pragma unroll
    for (int off = 16; off > 0; off >>= 1) {
        contrib += __shfl_down(contrib, off, 32);
        pcnt    += __shfl_down(pcnt,    off, 32);
    }
    if (lane == 0) rowval[row] = contrib / pcnt;
}

// Deterministic final reduction: ap_matrix = mean_i rowval, then 1 - sum(ap*label)/sum(label)
__global__ void tb_ap_finalize_kernel(const float* __restrict__ rowval,
                                      const float* __restrict__ label,
                                      float* __restrict__ out)
{
    __shared__ float apm[BBDIM];
    const int t = threadIdx.x;
    if (t < BBDIM) {
        float acc = 0.0f;
        for (int i = 0; i < FDIM; ++i) acc += rowval[t * FDIM + i];
        apm[t] = (acc / (float)FDIM) * label[t];
    }
    __syncthreads();
    if (t == 0) {
        float sa = 0.0f, sl = 0.0f;
        for (int q = 0; q < BBDIM; ++q) { sa += apm[q]; sl += label[q]; }
        out[0] = 1.0f - sa / sl;
    }
}

extern "C" void kernel_launch(void* const* d_in, const int* in_sizes, int n_in,
                              void* d_out, int out_size, void* d_ws, size_t ws_size,
                              hipStream_t stream) {
    (void)in_sizes; (void)n_in; (void)out_size; (void)ws_size;
    const float* sim = (const float*)d_in[0];
    const float* pos = (const float*)d_in[1];
    const float* neg = (const float*)d_in[2];
    const float* lab = (const float*)d_in[3];
    float* rowvals = (float*)d_ws;       // 4096 floats of scratch
    float* out     = (float*)d_out;

    tb_ap_rows_kernel<<<ROWS / WPB, WPB * 32, 0, stream>>>(sim, pos, neg, rowvals);
    tb_ap_finalize_kernel<<<1, 64, 0, stream>>>(rowvals, lab, out);
}
